// LocallyConnected2D_49005576847841
// MI455X (gfx1250) — matrix operand order repaired and checked by reading, not compiled, not run on hardware
//
#include <hip/hip_runtime.h>
#include <hip/hip_bf16.h>

typedef _Float16 v16h __attribute__((ext_vector_type(16)));
typedef float    v8f  __attribute__((ext_vector_type(8)));
typedef float    v4f  __attribute__((ext_vector_type(4)));

// Problem constants (from reference)
#define BB   16
#define HH   64
#define WW   64
#define CIN  32
#define FF   64
#define ORR  62
#define OCC  62
#define PP   (ORR * OCC)   // 3844
#define KK   288           // 3*3*32

__global__ __launch_bounds__(256)
void lc2d_wmma_kernel(const float* __restrict__ x,
                      const float* __restrict__ kern,
                      const float* __restrict__ bias,
                      float* __restrict__ out) {
    const int wave = threadIdx.x >> 5;
    const int lane = threadIdx.x & 31;
    const int lo   = lane & 15;       // lane within half-wave
    const int hi   = lane >> 4;       // which half-wave

    const int p = blockIdx.x * 8 + wave;   // one output position per wave
    if (p >= PP) return;                   // wave-uniform: EXEC all-ones below

    const int orow = p / OCC;
    const int ocol = p % OCC;

    // ---- A (patches) base address ------------------------------------------
    // A-matrix 16x32 f16 lane layout:
    //   lane lo      (hi=0): VGPR0-3 = K 0..7,   VGPR4-7 = K 16..23
    //   lane lo+16   (hi=1): VGPR0-3 = K 8..15,  VGPR4-7 = K 24..31
    // K-chunk t == tap (i,j), K-within-chunk == channel c (contiguous in x).
    // row M = batch = lo.
    const float* xbase =
        x + (((size_t)lo * HH + orow) * WW + ocol) * CIN + hi * 8;

    // ---- B (weights) base address ------------------------------------------
    // B-matrix 32x16 f16 lane layout: column N = f = nt*16 + lo;
    //   lanes 0-15: K 0..7, 16..23 ; lanes 16-31: K 8..15, 24..31.
    const float* kbase =
        kern + (size_t)p * KK * FF + (size_t)hi * 8 * FF + lo;

    v8f acc[4];
    acc[0] = (v8f){}; acc[1] = (v8f){}; acc[2] = (v8f){}; acc[3] = (v8f){};

    for (int i = 0; i < 3; ++i) {
        for (int j = 0; j < 3; ++j) {
            const int t = i * 3 + j;            // K-chunk index (0..8)

            // ---- load + convert A chunk (16 fp32 -> 16 f16 per lane) ----
            const float* xp = xbase + ((size_t)i * WW + j) * CIN;
            v4f x0 = *(const v4f*)(xp);
            v4f x1 = *(const v4f*)(xp + 4);
            v4f x2 = *(const v4f*)(xp + 16);
            v4f x3 = *(const v4f*)(xp + 20);

            v16h a;
            a[0]  = (_Float16)x0[0]; a[1]  = (_Float16)x0[1];
            a[2]  = (_Float16)x0[2]; a[3]  = (_Float16)x0[3];
            a[4]  = (_Float16)x1[0]; a[5]  = (_Float16)x1[1];
            a[6]  = (_Float16)x1[2]; a[7]  = (_Float16)x1[3];
            a[8]  = (_Float16)x2[0]; a[9]  = (_Float16)x2[1];
            a[10] = (_Float16)x2[2]; a[11] = (_Float16)x2[3];
            a[12] = (_Float16)x3[0]; a[13] = (_Float16)x3[1];
            a[14] = (_Float16)x3[2]; a[15] = (_Float16)x3[3];

            const float* kp = kbase + (size_t)t * 32 * FF;

            // ---- 4 N-tiles: load/convert B chunk, WMMA accumulate ----
#pragma unroll
            for (int nt = 0; nt < 4; ++nt) {
                const float* kq = kp + nt * 16;
                v16h bm;
#pragma unroll
                for (int e = 0; e < 16; ++e)
                    bm[e] = (_Float16)kq[(size_t)(e + (e >= 8 ? 8 : 0)) * FF];

                acc[nt] = __builtin_amdgcn_wmma_f32_16x16x32_f16(
                    /*neg_a=*/false, a, /*neg_b=*/false, bm,
                    /*c_mod=*/(short)0, acc[nt],
                    /*reuse_a=*/false, /*reuse_b=*/false);
            }
        }
    }

    // ---- epilogue: bias + store --------------------------------------------
    // C/D layout: VGPR r, lane -> out row M = r + 8*hi, col N = nt*16 + lo
#pragma unroll
    for (int nt = 0; nt < 4; ++nt) {
        const float bv = bias[(size_t)p * FF + nt * 16 + lo];
#pragma unroll
        for (int r = 0; r < 8; ++r) {
            const int b = r + hi * 8;
            out[(((size_t)b * PP + p) * FF) + nt * 16 + lo] = acc[nt][r] + bv;
        }
    }
}

extern "C" void kernel_launch(void* const* d_in, const int* in_sizes, int n_in,
                              void* d_out, int out_size, void* d_ws, size_t ws_size,
                              hipStream_t stream) {
    const float* x    = (const float*)d_in[0];
    const float* kern = (const float*)d_in[1];
    const float* bias = (const float*)d_in[2];
    float* out        = (float*)d_out;

    // 8 waves (positions) per 256-thread block
    const int blocks = (PP + 7) / 8;   // 481
    hipLaunchKernelGGL(lc2d_wmma_kernel, dim3(blocks), dim3(256), 0, stream,
                       x, kern, bias, out);
}
